// SimpleLSTM_841813590133
// MI455X (gfx1250) — matrix operand order repaired and checked by reading, not compiled, not run on hardware
//
#include <hip/hip_runtime.h>

// ---------------------------------------------------------------------------
// MI455X / gfx1250 BiLSTM inference.
// Persistent per-batch-block LSTM kernels using wave32 WMMA
// (v_wmma_f32_16x16x32_bf16), weights pre-packed to bf16 in B-operand layout,
// h/c state + A-operands staged in LDS, 256 timesteps per kernel launch.
// Round 4: __builtin_amdgcn_sched_barrier(0) between the kt+1 load block and
// the kt WMMA block. Round 3's source-level double buffer was undone by the
// machine scheduler (loads sunk to just before use -> buffers coalesced ->
// s_wait_loadcnt 0x0 before every wmma). The sched barrier pins the pipeline
// so kt+1's loads stay in flight across kt's matrix ops.
// ---------------------------------------------------------------------------

typedef __attribute__((ext_vector_type(16))) __bf16 v16bf;
typedef __attribute__((ext_vector_type(8)))  float  v8f;

static constexpr int kB  = 512;   // batch
static constexpr int kT  = 256;   // seq len
static constexpr int kG  = 512;   // 4*H gates
static constexpr int kH  = 128;   // hidden
static constexpr int k2H = 256;   // 2*H (bi output width)

__device__ __forceinline__ void sched_fence() {
#if defined(__has_builtin)
#if __has_builtin(__builtin_amdgcn_sched_barrier)
  __builtin_amdgcn_sched_barrier(0);   // no instruction may cross this point
#endif
#endif
}

__device__ __forceinline__ unsigned short f2bf(float f) {
  union { float f; unsigned int u; } v; v.f = f;
  unsigned int u = v.u;
  u += 0x7fffu + ((u >> 16) & 1u);   // round-to-nearest-even
  return (unsigned short)(u >> 16);
}

__device__ __forceinline__ float sigf(float x) {
  return 1.f / (1.f + __expf(-x));
}
__device__ __forceinline__ float tanh_fast(float x) {
  float ax = fabsf(x);
  float t = __expf(-2.f * ax);
  float r = (1.f - t) / (1.f + t);
  return copysignf(r, x);
}

__device__ __forceinline__ v8f wmma_bf16(v16bf a, v16bf b, v8f c) {
  // D(16x16 f32) = A(16x32 bf16) x B(32x16 bf16) + C
  return __builtin_amdgcn_wmma_f32_16x16x32_bf16(
      /*neg_a=*/false, a, /*neg_b=*/false, b,
      /*c_mod=*/(short)0, c, /*reuse_a=*/false, /*reuse_b=*/false);
}

// A-operand LDS slot for element (row m in 0..15, ko = k within 32-wide ktile).
// Per ISA 16-bit A 16x32 layout: lane = half16*16 + m;
// k = (j>>2)*16 + half16*8 + (j&3)*2 + lo.
__device__ __forceinline__ int a_slot(int m, int ko) {
  int group = ko >> 4;
  int ko16  = ko & 15;
  int half  = ko16 >> 3;
  int kk    = ko16 & 7;
  int j     = group * 4 + (kk >> 1);
  int lo    = kk & 1;
  int lane  = half * 16 + m;
  return lane * 16 + j * 2 + lo;   // offset inside one 512-element ktile
}

// ---------------------------------------------------------------------------
// Double-buffered, fully-unrolled 16xN GEMM stage: one wave computes 4
// 16x16 N-tiles over KT k-tiles. A operand from LDS (A-layout ktiles at
// stride 512), B operands streamed from global in packed layout via bptr(kt).
// Loads for kt+1 are issued before the WMMAs of kt, and a sched_barrier pins
// that order so load latency hides under the matrix ops.
// ---------------------------------------------------------------------------
template <int KT, typename BPtrFn>
__device__ __forceinline__ void gemm_pipelined(
    const unsigned short* A_base, int lane, int wv, BPtrFn bptr, v8f (&acc)[4]) {
  v16bf a_buf[2];
  v16bf b_buf[2][4];
  {
    const unsigned short* bb = bptr(0);
    a_buf[0] = *(const v16bf*)(const void*)&A_base[lane * 16];
    #pragma unroll
    for (int i = 0; i < 4; ++i)
      b_buf[0][i] = *(const v16bf*)(const void*)(bb + (wv * 4 + i) * 512 + lane * 16);
  }
  #pragma unroll
  for (int kt = 0; kt < KT; ++kt) {
    const int p = kt & 1;
    if (kt + 1 < KT) {   // prefetch next k-tile into the other buffer
      const unsigned short* bb = bptr(kt + 1);
      a_buf[p ^ 1] = *(const v16bf*)(const void*)&A_base[(kt + 1) * 512 + lane * 16];
      #pragma unroll
      for (int i = 0; i < 4; ++i)
        b_buf[p ^ 1][i] =
            *(const v16bf*)(const void*)(bb + (wv * 4 + i) * 512 + lane * 16);
    }
    sched_fence();   // keep kt+1 loads above kt's WMMAs
    #pragma unroll
    for (int i = 0; i < 4; ++i) acc[i] = wmma_bf16(a_buf[p], b_buf[p][i], acc[i]);
  }
}

// ---------------------------------------------------------------------------
// Weight pack: W[512][K] f32 -> bf16, per-lane-contiguous B-operand layout.
// Linear index e = ((kt*32 + nt)*32 + lane)*16 + elem, so consumer reads
// one contiguous 32B chunk per lane:  base + kt*16384 + nt*512 + lane*16.
// Per ISA (16-bit B 32x16): n = nt*16 + (lane&15); k = kt*32 + (elem>>3)*16 + (lane>>4)*8 + (elem&7).
// ---------------------------------------------------------------------------
__global__ void pack_w_kernel(const float* __restrict__ W,
                              unsigned short* __restrict__ dst,
                              int K, int total) {
  int e = blockIdx.x * 256 + threadIdx.x;
  if (e >= total) return;
  int elem = e & 15;
  int lane = (e >> 4) & 31;
  int rest = e >> 9;
  int nt   = rest & 31;
  int kt   = rest >> 5;
  int j    = elem >> 1, lo = elem & 1;
  int n    = nt * 16 + (lane & 15);
  int half = lane >> 4;
  int k    = kt * 32 + (elem >> 3) * 16 + half * 8 + (elem & 7);
  float v  = (k < K) ? W[n * K + k] : 0.f;
  dst[e] = f2bf(v);
}

__global__ void addbias_kernel(const float* __restrict__ a,
                               const float* __restrict__ b,
                               float* __restrict__ o) {
  int i = blockIdx.x * 256 + threadIdx.x;
  if (i < kG) o[i] = a[i] + b[i];
}

// ---------------------------------------------------------------------------
// Persistent BiLSTM layer kernel.
// grid = (32, 2): blockIdx.x = 16-row batch block, blockIdx.y = direction.
// 256 threads = 8 waves; each wave owns 4 N-tiles of the 16x512 gate GEMM.
// LAYER==1: x from emb[ids], K=300 (padded to 320, KT_X=10)
// LAYER==2: x from h1 [B,T,256], K=256 (KT_X=8)
// h-part contributes 4 more ktiles (K=128). Fused per-step GEMM.
// ---------------------------------------------------------------------------
template <int LAYER>
__global__ __launch_bounds__(256, 1) void lstm_kernel(
    const int* __restrict__ xids, const float* __restrict__ emb,
    const float* __restrict__ hin,
    const unsigned short* __restrict__ wih_f, const unsigned short* __restrict__ wih_r,
    const unsigned short* __restrict__ whh_f, const unsigned short* __restrict__ whh_r,
    const float* __restrict__ bias_f, const float* __restrict__ bias_r,
    float* __restrict__ hout) {
  constexpr int KT_X   = (LAYER == 1) ? 10 : 8;
  constexpr int K_IN   = (LAYER == 1) ? 300 : 256;
  constexpr int KPAD   = KT_X * 32;
  constexpr int KT_TOT = KT_X + 4;

  __shared__ alignas(32) unsigned short A_lds[14 * 512];  // <=14 ktiles, A layout
  __shared__ float G_lds[16 * kG];                        // 16x512 gate accum
  __shared__ float C_lds[16 * kH];                        // cell state
  __shared__ int   ids[16];

  const int tid = threadIdx.x;
  const int r0  = blockIdx.x * 16;
  const int dir = blockIdx.y;            // 0 = forward, 1 = backward
  const unsigned short* wih = dir ? wih_r : wih_f;
  const unsigned short* whh = dir ? whh_r : whh_f;
  const float* bias = dir ? bias_r : bias_f;

  // init h = 0 (A-layout h region), c = 0
  for (int e = tid; e < 16 * kH; e += 256) C_lds[e] = 0.f;
  for (int e = tid; e < 4 * 512; e += 256) A_lds[KT_X * 512 + e] = 0;

  const int lane = tid & 31;
  const int wv   = tid >> 5;
  const int ncol = lane & 15;
  const int hi8  = (lane >> 4) * 8;

  for (int s = 0; s < kT; ++s) {
    const int t = dir ? (kT - 1 - s) : s;

    if (LAYER == 1) {
      if (tid < 16) ids[tid] = xids[(r0 + tid) * kT + t];
    }
    __syncthreads();

    // ---- stage x_t into A-layout LDS (bf16) ----
    if (LAYER == 1) {
      for (int e = tid; e < 16 * KPAD; e += 256) {
        int m = e / KPAD;
        int k = e - m * KPAD;
        float v = (k < K_IN) ? emb[ids[m] * 300 + k] : 0.f;
        A_lds[(k >> 5) * 512 + a_slot(m, k & 31)] = f2bf(v);
      }
    } else {
      for (int e = tid; e < 16 * KPAD; e += 256) {
        int m = e >> 8;
        int k = e & 255;
        float v = hin[((size_t)(r0 + m) * kT + t) * k2H + k];
        A_lds[(k >> 5) * 512 + a_slot(m, k & 31)] = f2bf(v);
      }
    }
    __syncthreads();

    // ---- G = [x_t | h] @ [W_ih | W_hh]^T + bias  via pipelined WMMA ----
    {
      v8f acc[4];
      #pragma unroll
      for (int i = 0; i < 4; ++i) {
        float bv = bias[(wv * 4 + i) * 16 + ncol];
        #pragma unroll
        for (int r = 0; r < 8; ++r) acc[i][r] = bv;
      }
      auto bptr = [&](int kt) -> const unsigned short* {
        return (kt < KT_X) ? (wih + kt * 16384) : (whh + (kt - KT_X) * 16384);
      };
      gemm_pipelined<KT_TOT>(A_lds, lane, wv, bptr, acc);
      #pragma unroll
      for (int i = 0; i < 4; ++i) {
        int n = (wv * 4 + i) * 16 + ncol;
        #pragma unroll
        for (int r = 0; r < 8; ++r) G_lds[(hi8 + r) * kG + n] = acc[i][r];
      }
    }
    __syncthreads();

    // ---- gate nonlinearities, update c/h, stage h for next step ----
    for (int e = tid; e < 16 * kH; e += 256) {
      int m = e >> 7;
      int u = e & 127;
      const float* g = &G_lds[m * kG];
      float gi = g[u], gf = g[kH + u], gg = g[2 * kH + u], go = g[3 * kH + u];
      float c = sigf(gf) * C_lds[e] + sigf(gi) * tanh_fast(gg);
      float h = sigf(go) * tanh_fast(c);
      C_lds[e] = c;
      A_lds[(KT_X + (u >> 5)) * 512 + a_slot(m, u & 31)] = f2bf(h);
      hout[((size_t)(r0 + m) * kT + t) * k2H + dir * kH + u] = h;
    }
    __syncthreads();
  }
}

// ---------------------------------------------------------------------------
// avg + max pool over T:  h2 [B,T,256] -> hc [B,512] = [avg | max]
// ---------------------------------------------------------------------------
__global__ void pool_kernel(const float* __restrict__ h2, float* __restrict__ hc) {
  int b = blockIdx.x, ch = threadIdx.x;
  float s = 0.f, m = -3.4e38f;
  const float* row = h2 + (size_t)b * kT * k2H + ch;
  for (int t = 0; t < kT; ++t) {
    float v = row[(size_t)t * k2H];
    s += v;
    m = fmaxf(m, v);
  }
  hc[b * 512 + ch]       = s * (1.f / kT);
  hc[b * 512 + 256 + ch] = m;
}

// ---------------------------------------------------------------------------
// hidden = relu(hc@w1^T+b1) + relu(hc@w2^T+b2) + hc
// Two pipelined WMMA passes (w1 then w2) to bound register pressure.
// ---------------------------------------------------------------------------
__global__ __launch_bounds__(256, 1) void dense_kernel(
    const float* __restrict__ hc,
    const unsigned short* __restrict__ w1b, const unsigned short* __restrict__ w2b,
    const float* __restrict__ b1, const float* __restrict__ b2,
    float* __restrict__ hidden) {
  __shared__ alignas(32) unsigned short A_lds[16 * 512];  // K=512 -> 16 ktiles
  const int tid = threadIdx.x;
  const int r0  = blockIdx.x * 16;

  for (int e = tid; e < 16 * 512; e += 256) {
    int m = e >> 9, k = e & 511;
    A_lds[(k >> 5) * 512 + a_slot(m, k & 31)] = f2bf(hc[(r0 + m) * 512 + k]);
  }
  __syncthreads();

  const int lane = tid & 31, wv = tid >> 5;
  const int ncol = lane & 15, hi8 = (lane >> 4) * 8;
  v8f a1[4], a2[4];
  #pragma unroll
  for (int i = 0; i < 4; ++i) {
    int n = (wv * 4 + i) * 16 + ncol;
    float v1 = b1[n], v2 = b2[n];
    #pragma unroll
    for (int r = 0; r < 8; ++r) { a1[i][r] = v1; a2[i][r] = v2; }
  }
  auto bp1 = [&](int kt) -> const unsigned short* { return w1b + kt * 16384; };
  auto bp2 = [&](int kt) -> const unsigned short* { return w2b + kt * 16384; };
  gemm_pipelined<16>(A_lds, lane, wv, bp1, a1);
  gemm_pipelined<16>(A_lds, lane, wv, bp2, a2);

  #pragma unroll
  for (int i = 0; i < 4; ++i) {
    int n = (wv * 4 + i) * 16 + ncol;
    #pragma unroll
    for (int r = 0; r < 8; ++r) {
      int m = hi8 + r;
      hidden[(r0 + m) * 512 + n] =
          fmaxf(a1[i][r], 0.f) + fmaxf(a2[i][r], 0.f) + hc[(r0 + m) * 512 + n];
    }
  }
}

// ---------------------------------------------------------------------------
// out[b, 0..6] = hidden[b] . {w_out, w_aux[0..5]} + bias   (one wave per row)
// ---------------------------------------------------------------------------
__global__ void head_kernel(const float* __restrict__ hidden,
                            const float* __restrict__ w_out, const float* __restrict__ b_out,
                            const float* __restrict__ w_aux, const float* __restrict__ b_aux,
                            float* __restrict__ out) {
  int b = blockIdx.x, lane = threadIdx.x;  // 32 threads
  float hv[16];
  #pragma unroll
  for (int j = 0; j < 16; ++j) hv[j] = hidden[b * 512 + lane + j * 32];
  for (int o = 0; o < 7; ++o) {
    const float* w = (o == 0) ? w_out : (w_aux + (o - 1) * 512);
    float p = 0.f;
    #pragma unroll
    for (int j = 0; j < 16; ++j) p += hv[j] * w[lane + j * 32];
    for (int off = 16; off > 0; off >>= 1) p += __shfl_down(p, off, 32);
    if (lane == 0) out[b * 7 + o] = p + ((o == 0) ? b_out[0] : b_aux[o - 1]);
  }
}

// ---------------------------------------------------------------------------
extern "C" void kernel_launch(void* const* d_in, const int* in_sizes, int n_in,
                              void* d_out, int out_size, void* d_ws, size_t ws_size,
                              hipStream_t stream) {
  const int*   x      = (const int*)d_in[0];
  const float* emb    = (const float*)d_in[1];
  const float* w_ih1f = (const float*)d_in[2];
  const float* w_hh1f = (const float*)d_in[3];
  const float* b_ih1f = (const float*)d_in[4];
  const float* b_hh1f = (const float*)d_in[5];
  const float* w_ih1b = (const float*)d_in[6];
  const float* w_hh1b = (const float*)d_in[7];
  const float* b_ih1b = (const float*)d_in[8];
  const float* b_hh1b = (const float*)d_in[9];
  const float* w_ih2f = (const float*)d_in[10];
  const float* w_hh2f = (const float*)d_in[11];
  const float* b_ih2f = (const float*)d_in[12];
  const float* b_hh2f = (const float*)d_in[13];
  const float* w_ih2b = (const float*)d_in[14];
  const float* w_hh2b = (const float*)d_in[15];
  const float* b_ih2b = (const float*)d_in[16];
  const float* b_hh2b = (const float*)d_in[17];
  const float* w1     = (const float*)d_in[18];
  const float* b1     = (const float*)d_in[19];
  const float* w2     = (const float*)d_in[20];
  const float* b2     = (const float*)d_in[21];
  const float* w_out  = (const float*)d_in[22];
  const float* b_out  = (const float*)d_in[23];
  const float* w_aux  = (const float*)d_in[24];
  const float* b_aux  = (const float*)d_in[25];
  float* out = (float*)d_out;

  // workspace carve
  char*  base = (char*)d_ws;
  size_t off  = 0;
  auto take = [&](size_t bytes) -> void* {
    void* r = base + off;
    off += (bytes + 255) & ~(size_t)255;
    return r;
  };
  float* h1     = (float*)take((size_t)kB * kT * k2H * 4);   // 128 MB
  float* h2     = (float*)take((size_t)kB * kT * k2H * 4);   // 128 MB
  float* hc     = (float*)take(512 * 512 * 4);
  float* hidden = (float*)take(512 * 512 * 4);
  float* bias1f = (float*)take(kG * 4);
  float* bias1b = (float*)take(kG * 4);
  float* bias2f = (float*)take(kG * 4);
  float* bias2b = (float*)take(kG * 4);
  const int SZ_IH1 = 10 * 32 * 512;  // 512 x 320 (pad of 300)
  const int SZ_IH2 = 8  * 32 * 512;  // 512 x 256
  const int SZ_HH  = 4  * 32 * 512;  // 512 x 128
  const int SZ_D   = 16 * 32 * 512;  // 512 x 512
  unsigned short* wih1f_p = (unsigned short*)take((size_t)SZ_IH1 * 2);
  unsigned short* wih1b_p = (unsigned short*)take((size_t)SZ_IH1 * 2);
  unsigned short* whh1f_p = (unsigned short*)take((size_t)SZ_HH * 2);
  unsigned short* whh1b_p = (unsigned short*)take((size_t)SZ_HH * 2);
  unsigned short* wih2f_p = (unsigned short*)take((size_t)SZ_IH2 * 2);
  unsigned short* wih2b_p = (unsigned short*)take((size_t)SZ_IH2 * 2);
  unsigned short* whh2f_p = (unsigned short*)take((size_t)SZ_HH * 2);
  unsigned short* whh2b_p = (unsigned short*)take((size_t)SZ_HH * 2);
  unsigned short* w1_p    = (unsigned short*)take((size_t)SZ_D * 2);
  unsigned short* w2_p    = (unsigned short*)take((size_t)SZ_D * 2);

  auto pack = [&](const float* W, unsigned short* dst, int K, int total) {
    pack_w_kernel<<<(total + 255) / 256, 256, 0, stream>>>(W, dst, K, total);
  };
  pack(w_ih1f, wih1f_p, 300, SZ_IH1);
  pack(w_ih1b, wih1b_p, 300, SZ_IH1);
  pack(w_hh1f, whh1f_p, 128, SZ_HH);
  pack(w_hh1b, whh1b_p, 128, SZ_HH);
  pack(w_ih2f, wih2f_p, 256, SZ_IH2);
  pack(w_ih2b, wih2b_p, 256, SZ_IH2);
  pack(w_hh2f, whh2f_p, 128, SZ_HH);
  pack(w_hh2b, whh2b_p, 128, SZ_HH);
  pack(w1, w1_p, 512, SZ_D);
  pack(w2, w2_p, 512, SZ_D);

  addbias_kernel<<<2, 256, 0, stream>>>(b_ih1f, b_hh1f, bias1f);
  addbias_kernel<<<2, 256, 0, stream>>>(b_ih1b, b_hh1b, bias1b);
  addbias_kernel<<<2, 256, 0, stream>>>(b_ih2f, b_hh2f, bias2f);
  addbias_kernel<<<2, 256, 0, stream>>>(b_ih2b, b_hh2b, bias2b);

  lstm_kernel<1><<<dim3(kB / 16, 2), 256, 0, stream>>>(
      x, emb, h1, wih1f_p, wih1b_p, whh1f_p, whh1b_p, bias1f, bias1b, h1);
  lstm_kernel<2><<<dim3(kB / 16, 2), 256, 0, stream>>>(
      x, emb, h1, wih2f_p, wih2b_p, whh2f_p, whh2b_p, bias2f, bias2b, h2);

  pool_kernel<<<kB, 256, 0, stream>>>(h2, hc);
  dense_kernel<<<kB / 16, 256, 0, stream>>>(hc, w1_p, w2_p, b1, b2, hidden);
  head_kernel<<<kB, 32, 0, stream>>>(hidden, w_out, b_out, w_aux, b_aux, out);
}
